// SelfDCA_50594714746893
// MI455X (gfx1250) — matrix operand order repaired and checked by reading, not compiled, not run on hardware
//
#include <hip/hip_runtime.h>
#include <hip/hip_fp16.h>

#define NB      2       // batch
#define NDIM    256     // channels
#define NHEADS  8
#define DHEAD   32
#define NHW     4096    // 64*64
#define NKV     1024    // 32*32 sampled points per head
#define NBG     16      // NB*NHEADS

typedef __attribute__((ext_vector_type(16))) _Float16 v16h;
typedef __attribute__((ext_vector_type(8)))  _Float16 v8h;
typedef __attribute__((ext_vector_type(8)))  float    v8f;

union AFrag { v16h v; v8h h[2]; };

__device__ __forceinline__ v8f wmma_f16(v16h a, v16h b, v8f c) {
    // D = A(16x32 f16) * B(32x16 f16) + C(16x16 f32)
    return __builtin_amdgcn_wmma_f32_16x16x32_f16(false, a, false, b, (short)0, c,
                                                  false, false);
}

// ---------------------------------------------------------------------------
// x (B,C,H,W) f32  ->  x_t (B*HW, C) f16   (A-matrix, row-major over K=C)
// ---------------------------------------------------------------------------
__global__ void k_pack_x(const float* __restrict__ x, _Float16* __restrict__ xt) {
    int tid = blockIdx.x * blockDim.x + threadIdx.x;
    if (tid >= NB * NHW * NDIM) return;
    int c  = tid & (NDIM - 1);
    int r  = tid >> 8;
    int hw = r & (NHW - 1);
    int b  = r >> 12;
    xt[tid] = (_Float16)x[((size_t)(b * NDIM + c)) * NHW + hw];
}

// ---------------------------------------------------------------------------
// Generic B-fragment swizzle: B[k][n] = src[k*strideK + n*strideN]
// dst fragment layout (wave32, per ISA 7.12.2): frag = nt*Ksteps + kst,
// element e of lane L:  K = kst*32 + (e/8)*16 + (L/16)*8 + e%8 ;  N = nt*16 + (L%16)
// one frag = 32 lanes * 16 halves = 512 halves (1 KB), lane chunk is 32B.
// ---------------------------------------------------------------------------
__global__ void k_pack_b_f32(const float* __restrict__ src, _Float16* __restrict__ dst,
                             int Ksteps, int Ntiles, int strideK, int strideN,
                             size_t srcBS, size_t dstBS) {
    int batch = blockIdx.y;
    int tid = blockIdx.x * blockDim.x + threadIdx.x;
    if (tid >= Ksteps * Ntiles * 512) return;
    int e    = tid & 15;
    int lane = (tid >> 4) & 31;
    int frag = tid >> 9;
    int kst  = frag % Ksteps;
    int nt   = frag / Ksteps;
    int k = kst * 32 + ((e >> 3) << 4) + ((lane >> 4) << 3) + (e & 7);
    int n = nt * 16 + (lane & 15);
    dst[batch * dstBS + (size_t)frag * 512 + lane * 16 + e] =
        (_Float16)src[batch * srcBS + (size_t)k * strideK + (size_t)n * strideN];
}

__global__ void k_pack_b_f16(const _Float16* __restrict__ src, _Float16* __restrict__ dst,
                             int Ksteps, int Ntiles, int strideK, int strideN,
                             size_t srcBS, size_t dstBS) {
    int batch = blockIdx.y;
    int tid = blockIdx.x * blockDim.x + threadIdx.x;
    if (tid >= Ksteps * Ntiles * 512) return;
    int e    = tid & 15;
    int lane = (tid >> 4) & 31;
    int frag = tid >> 9;
    int kst  = frag % Ksteps;
    int nt   = frag / Ksteps;
    int k = kst * 32 + ((e >> 3) << 4) + ((lane >> 4) << 3) + (e & 7);
    int n = nt * 16 + (lane & 15);
    dst[batch * dstBS + (size_t)frag * 512 + lane * 16 + e] =
        src[batch * srcBS + (size_t)k * strideK + (size_t)n * strideN];
}

// ---------------------------------------------------------------------------
// WMMA GEMM  D[M x 256] = A[M x 256] * Bsw + bias, one 16x16 tile per wave.
// A: row-major f16 ld=256.  Bsw: pre-swizzled fragments (16 ntiles x 8 ksteps).
// mode 0: q projection  -> q_f32 (B,C,HW)  and q_t f16 (BG,HW,32) * scale
// mode 1: k/v projection-> f16 (BG,32,NKV)
// mode 2: final wo      -> outF f32 (B,C,HW)
// ---------------------------------------------------------------------------
__global__ void k_gemm(const _Float16* __restrict__ A, const _Float16* __restrict__ Bsw,
                       const float* __restrict__ bias, int tilesM, int mode, float scale,
                       float* __restrict__ outF, _Float16* __restrict__ outH) {
    int lane = threadIdx.x & 31;
    int wave = threadIdx.x >> 5;
    int tile = blockIdx.x * 8 + wave;               // blockDim.x == 256 (8 waves)
    if (tile >= tilesM * 16) return;
    int mt = tile % tilesM;
    int nt = tile / tilesM;
    int half = lane >> 4, mrow = lane & 15, col = lane & 15;

    const _Float16* arow  = A + ((size_t)(mt * 16 + mrow)) * 256 + half * 8;
    const _Float16* bbase = Bsw + ((size_t)nt * 8) * 512 + lane * 16;

    v8f acc = {};
#pragma unroll
    for (int ks = 0; ks < 8; ++ks) {
        AFrag a;
        a.h[0] = *(const v8h*)(arow + ks * 32);        // K = ks*32 + 8h .. +8
        a.h[1] = *(const v8h*)(arow + ks * 32 + 16);   // K = ks*32 + 16+8h ..
        v16h b = *(const v16h*)(bbase + ks * 512);
        if (ks + 1 < 8) __builtin_prefetch(bbase + (ks + 1) * 512, 0, 1);
        acc = wmma_f16(a.v, b, acc);
    }

    int n  = nt * 16 + col;
    float bn = bias ? bias[n] : 0.f;
#pragma unroll
    for (int v = 0; v < 8; ++v) {
        int m = mt * 16 + v + 8 * half;                // C/D layout: M = v + 8*(L/16)
        float val = acc[v] + bn;
        if (mode == 0) {
            int b  = m >> 12, hw = m & 4095;
            outF[((size_t)(b * NDIM + n)) * NHW + hw] = val;
            int bg = b * NHEADS + (n >> 5), ch = n & 31;
            outH[((size_t)bg * NHW + hw) * DHEAD + ch] = (_Float16)(val * scale);
        } else if (mode == 1) {
            int b  = m >> 10, pos = m & 1023;
            int bg = b * NHEADS + (n >> 5), ch = n & 31;
            outH[((size_t)(bg * DHEAD + ch)) * NKV + pos] = (_Float16)val;
        } else {
            int b = m >> 12, hw = m & 4095;
            outF[((size_t)(b * NDIM + n)) * NHW + hw] = val;
        }
    }
}

// ---------------------------------------------------------------------------
// Offset branch + bilinear sampling: one wave per (bg, sample point),
// lane = channel (DHEAD==32==wave32). dwconv5x5/s2 -> LN(32) -> GELU ->
// 1x1 to (dy,dx) -> +ref, clip -> bilinear sample x -> kv_t (B*NKV, 256) f16
// ---------------------------------------------------------------------------
__global__ void k_offset_sample(const float* __restrict__ x, const float* __restrict__ q,
                                const float* __restrict__ w_dw, const float* __restrict__ b_dw,
                                const float* __restrict__ ln_g, const float* __restrict__ ln_b,
                                const float* __restrict__ w_off, _Float16* __restrict__ kvt) {
    int lane = threadIdx.x & 31;
    int wid  = blockIdx.x * (blockDim.x >> 5) + (threadIdx.x >> 5);
    if (wid >= NBG * NKV) return;
    int bg = wid >> 10, pos = wid & 1023;
    int b = bg >> 3, head = bg & 7;
    int ky = pos >> 5, kx = pos & 31;

    // depthwise 5x5, stride 2, pad 2, on q (per-head view, channel=lane)
    const float* qc = q + ((size_t)(b * NDIM + head * DHEAD + lane)) * NHW;
    float s = b_dw[lane];
    int iy0 = ky * 2 - 2, ix0 = kx * 2 - 2;
#pragma unroll
    for (int dy = 0; dy < 5; ++dy) {
        int yy = iy0 + dy;
        if (yy < 0 || yy >= 64) continue;
#pragma unroll
        for (int dx = 0; dx < 5; ++dx) {
            int xx = ix0 + dx;
            if (xx < 0 || xx >= 64) continue;
            s += qc[yy * 64 + xx] * w_dw[lane * 25 + dy * 5 + dx];
        }
    }
    // LayerNorm over 32 channels (full-wave shfl_xor reduction)
    float mu = s;
    for (int m = 16; m; m >>= 1) mu += __shfl_xor(mu, m, 32);
    mu *= (1.f / 32.f);
    float d = s - mu;
    float var = d * d;
    for (int m = 16; m; m >>= 1) var += __shfl_xor(var, m, 32);
    var *= (1.f / 32.f);
    float t = d * __frsqrt_rn(var + 1e-5f) * ln_g[lane] + ln_b[lane];
    t = 0.5f * t * (1.f + erff(t * 0.70710678118654752f));   // exact GELU
    // 1x1 -> (dy,dx)
    float oy = t * w_off[lane];
    float ox = t * w_off[32 + lane];
    for (int m = 16; m; m >>= 1) {
        oy += __shfl_xor(oy, m, 32);
        ox += __shfl_xor(ox, m, 32);
    }
    float ry = (0.5f + (float)ky) * (2.f / 31.f) - 1.f;
    float rx = (0.5f + (float)kx) * (2.f / 31.f) - 1.f;
    float py = fminf(fmaxf(oy + ry, -1.f), 1.f);
    float px = fminf(fmaxf(ox + rx, -1.f), 1.f);
    // bilinear (align_corners=True)
    float fxp = (px + 1.f) * 0.5f * 63.f;
    float fyp = (py + 1.f) * 0.5f * 63.f;
    float x0 = floorf(fxp), y0 = floorf(fyp);
    float fx = fxp - x0,    fy = fyp - y0;
    int x0i = min(max((int)x0, 0), 63);
    int y0i = min(max((int)y0, 0), 63);
    int x1i = min(x0i + 1, 63), y1i = min(y0i + 1, 63);
    const float* img = x + ((size_t)(b * NDIM + head * DHEAD + lane)) * NHW;
    float v00 = img[y0i * 64 + x0i], v01 = img[y0i * 64 + x1i];
    float v10 = img[y1i * 64 + x0i], v11 = img[y1i * 64 + x1i];
    float val = v00 * (1.f - fx) * (1.f - fy) + v01 * fx * (1.f - fy) +
                v10 * (1.f - fx) * fy + v11 * fx * fy;
    kvt[((size_t)(b * NKV + pos)) * NDIM + head * DHEAD + lane] = (_Float16)val;
}

// ---------------------------------------------------------------------------
// Fused attention: per wave, a 16-row strip of one head group.
//   S(16x1024 f32) lives in LDS (320KB/WGP makes this free at 2 waves/WG),
//   scores WMMA'd against pre-swizzled K frags, 2-pass softmax in LDS,
//   P re-read as f16 A-frags, out = P*V accumulated with WMMA.
// ---------------------------------------------------------------------------
__global__ void k_attn(const _Float16* __restrict__ qt, const _Float16* __restrict__ kB,
                       const _Float16* __restrict__ vB, _Float16* __restrict__ attnT) {
    extern __shared__ __align__(128) char smem[];
    int lane = threadIdx.x & 31;
    int wave = threadIdx.x >> 5;           // blockDim.x == 64 (2 waves)
    int bg = blockIdx.y;
    int m0 = (blockIdx.x * 2 + wave) * 16;
    int half = lane >> 4, mrow = lane & 15, col = lane & 15;
    int b = bg >> 3, head = bg & 7;

    float*    S  = (float*)(smem) + (size_t)wave * 16 * 1024;               // 2*64KB
    _Float16* P  = (_Float16*)(smem + 131072) + (size_t)wave * 16 * 1024;   // 2*32KB
    float*    RS = (float*)(smem + 196608) + wave * 16;                     // 2*64B

    // q A-fragment (K = 32 = one WMMA step), scale already folded in
    const _Float16* qb = qt + ((size_t)bg * NHW + m0 + mrow) * DHEAD + half * 8;
    AFrag a;
    a.h[0] = *(const v8h*)(qb);
    a.h[1] = *(const v8h*)(qb + 16);

    // scores: 64 N-tiles of 16, one WMMA each
    const _Float16* kf = kB + (size_t)bg * 64 * 512 + lane * 16;
    for (int nt = 0; nt < 64; ++nt) {
        v16h bf = *(const v16h*)(kf + nt * 512);
        v8f c = {};
        c = wmma_f16(a.v, bf, c);
#pragma unroll
        for (int v = 0; v < 8; ++v)
            S[(v + 8 * half) * 1024 + nt * 16 + col] = c[v];
    }
    __syncthreads();

    // softmax over N=1024: two lanes per row (512 cols each)
    {
        int row = lane >> 1, part = lane & 1;
        const float* Sr = S + row * 1024 + part * 512;
        _Float16*    Pr = P + row * 1024 + part * 512;
        float mx = -3.0e38f;
        for (int i = 0; i < 512; ++i) mx = fmaxf(mx, Sr[i]);
        mx = fmaxf(mx, __shfl_xor(mx, 1, 32));
        float sum = 0.f;
        for (int i = 0; i < 512; ++i) {
            float p = __expf(Sr[i] - mx);
            Pr[i] = (_Float16)p;
            sum += p;
        }
        sum += __shfl_xor(sum, 1, 32);
        if (!part) RS[row] = 1.f / sum;
    }
    __syncthreads();

    // out = P(16x1024) * V^T(1024x32): 32 K-steps, 2 N-tiles of channels
    v8f acc0 = {}, acc1 = {};
    const _Float16* Pb = P + (size_t)mrow * 1024 + half * 8;
    const _Float16* vf = vB + (size_t)bg * 64 * 512 + lane * 16;
    for (int kt = 0; kt < 32; ++kt) {
        AFrag pa;
        pa.h[0] = *(const v8h*)(Pb + kt * 32);
        pa.h[1] = *(const v8h*)(Pb + kt * 32 + 16);
        v16h b0 = *(const v16h*)(vf + (0 * 32 + kt) * 512);
        v16h b1 = *(const v16h*)(vf + (1 * 32 + kt) * 512);
        acc0 = wmma_f16(pa.v, b0, acc0);
        acc1 = wmma_f16(pa.v, b1, acc1);
    }
#pragma unroll
    for (int v = 0; v < 8; ++v) {
        int m = v + 8 * half;
        float inv = RS[m];
        size_t base = ((size_t)b * NHW + m0 + m) * NDIM + head * DHEAD;
        attnT[base + col]      = (_Float16)(acc0[v] * inv);
        attnT[base + 16 + col] = (_Float16)(acc1[v] * inv);
    }
}

// ---------------------------------------------------------------------------
extern "C" void kernel_launch(void* const* d_in, const int* in_sizes, int n_in,
                              void* d_out, int out_size, void* d_ws, size_t ws_size,
                              hipStream_t stream) {
    (void)in_sizes; (void)n_in; (void)out_size; (void)ws_size;
    const float* x    = (const float*)d_in[0];
    const float* wq   = (const float*)d_in[1];
    const float* bq   = (const float*)d_in[2];
    const float* wk   = (const float*)d_in[3];
    const float* bk   = (const float*)d_in[4];
    const float* wv   = (const float*)d_in[5];
    const float* bv   = (const float*)d_in[6];
    const float* wdw  = (const float*)d_in[7];
    const float* bdw  = (const float*)d_in[8];
    const float* lng  = (const float*)d_in[9];
    const float* lnb  = (const float*)d_in[10];
    const float* woff = (const float*)d_in[11];
    const float* wo   = (const float*)d_in[12];
    const float* bo   = (const float*)d_in[13];
    float* out = (float*)d_out;

    char* ws = (char*)d_ws;
    size_t off = 0;
    auto alloc = [&](size_t bytes) -> char* {
        char* p = ws + off;
        off += (bytes + 255) & ~(size_t)255;
        return p;
    };
    _Float16* xt    = (_Float16*)alloc((size_t)NB * NHW * NDIM * 2);     // 4 MB
    _Float16* wqB   = (_Float16*)alloc(16 * 8 * 512 * 2);                // 128 KB each
    _Float16* wkB   = (_Float16*)alloc(16 * 8 * 512 * 2);
    _Float16* wvB   = (_Float16*)alloc(16 * 8 * 512 * 2);
    _Float16* woB   = (_Float16*)alloc(16 * 8 * 512 * 2);
    float*    qf    = (float*)   alloc((size_t)NB * NDIM * NHW * 4);     // 8 MB
    _Float16* qt    = (_Float16*)alloc((size_t)NBG * NHW * DHEAD * 2);   // 4 MB
    _Float16* kvt   = (_Float16*)alloc((size_t)NB * NKV * NDIM * 2);     // 1 MB
    _Float16* kf16  = (_Float16*)alloc((size_t)NBG * DHEAD * NKV * 2);   // 1 MB
    _Float16* vf16  = (_Float16*)alloc((size_t)NBG * DHEAD * NKV * 2);   // 1 MB
    _Float16* kB    = (_Float16*)alloc((size_t)NBG * 64 * 512 * 2);      // 1 MB
    _Float16* vBs   = (_Float16*)alloc((size_t)NBG * 64 * 512 * 2);      // 1 MB
    _Float16* attnT = (_Float16*)alloc((size_t)NB * NHW * NDIM * 2);     // 4 MB

    const float scale = 0.17677669529663687f;  // 32^-0.5, folded into q_t

    // x -> f16 A-matrix layout
    k_pack_x<<<(NB * NHW * NDIM + 255) / 256, 256, 0, stream>>>(x, xt);

    // weight B-fragments: B[k=c][n=o] = w[o*256+c]  (strideK=1, strideN=256)
    k_pack_b_f32<<<dim3(256, 1), 256, 0, stream>>>(wq, wqB, 8, 16, 1, 256, 0, 0);
    k_pack_b_f32<<<dim3(256, 1), 256, 0, stream>>>(wk, wkB, 8, 16, 1, 256, 0, 0);
    k_pack_b_f32<<<dim3(256, 1), 256, 0, stream>>>(wv, wvB, 8, 16, 1, 256, 0, 0);
    k_pack_b_f32<<<dim3(256, 1), 256, 0, stream>>>(wo, woB, 8, 16, 1, 256, 0, 0);

    // q projection (M=8192): writes q f32 (for dwconv) + scaled q_t f16
    k_gemm<<<1024, 256, 0, stream>>>(xt, wqB, bq, 512, 0, scale, qf, qt);

    // offset branch + bilinear gather -> kv_t (M=2048, K=256) f16
    k_offset_sample<<<4096, 128, 0, stream>>>(x, qf, wdw, bdw, lng, lnb, woff, kvt);

    // k / v projections (M=2048) -> (BG,32,1024) f16
    k_gemm<<<256, 256, 0, stream>>>(kvt, wkB, bk, 128, 1, 1.f, nullptr, kf16);
    k_gemm<<<256, 256, 0, stream>>>(kvt, wvB, bv, 128, 1, 1.f, nullptr, vf16);

    // K as B-frags:  B[k=c][n=pos], per group (Ksteps=1, Ntiles=64)
    k_pack_b_f16<<<dim3(128, NBG), 256, 0, stream>>>(kf16, kB, 1, 64, 1024, 1,
                                                     (size_t)DHEAD * NKV, 64 * 512);
    // V^T as B-frags: B[k=pos][n=c], per group (Ksteps=32, Ntiles=2)
    k_pack_b_f16<<<dim3(128, NBG), 256, 0, stream>>>(vf16, vBs, 32, 2, 1, 1024,
                                                     (size_t)DHEAD * NKV, 64 * 512);

    // fused scores -> softmax(LDS) -> P*V; 2 waves/WG, ~192KB dynamic LDS
    size_t lds = 131072 + 65536 + 256;  // S(2x64K) + P(2x32K) + row sums
    k_attn<<<dim3(128, NBG), 64, lds, stream>>>(qt, kB, vBs, attnT);

    // final wo projection -> d_out f32 (B,256,64,64)
    k_gemm<<<1024, 256, 0, stream>>>(attnT, woB, bo, 512, 2, 1.f, out, nullptr);
}
